// Model_39676907887018
// MI455X (gfx1250) — matrix operand order repaired and checked by reading, not compiled, not run on hardware
//
#include <hip/hip_runtime.h>
#include <hip/hip_bf16.h>

// ---------------------------------------------------------------------------
// Attention pipeline for MI455X (gfx1250, wave32, WMMA), compile-time
// specialized GEMMs:
//   1) Q/K/V projections  : bf16 WMMA GEMM + bias, fp32 in -> bf16 out
//   2) S = (Q K^T) * 8    : bf16 WMMA GEMM (NT), fp32 out
//   3) softmax + dropout  : VALU kernel, fp32 in -> bf16 P
//   4) O = P V            : bf16 WMMA GEMM (NN), fp32 out
// Block tile 128x256 (8 waves), wave tile 64x64 -> 16 v_wmma per 32-K step,
// each A/B fragment reused 4x.
// ---------------------------------------------------------------------------

typedef __attribute__((ext_vector_type(16))) __bf16 v16bf;
typedef __attribute__((ext_vector_type(8)))  float  v8f;

union Frag16 { v16bf v; unsigned int u[8]; };
union FragC  { v8f v; float f[8]; };

__device__ __forceinline__ unsigned short f2bf(float x) {
    unsigned int u = __float_as_uint(x);
    unsigned int r = u + 0x7FFFu + ((u >> 16) & 1u);   // round-to-nearest-even
    return (unsigned short)(r >> 16);
}
// two f32 -> packed bf16 pair: 2x v_add3_u32 + 1x v_perm_b32
__device__ __forceinline__ unsigned int pack2bf(float x, float y) {
    unsigned int ux = __float_as_uint(x);
    unsigned int uy = __float_as_uint(y);
    ux += 0x7FFFu + ((ux >> 16) & 1u);
    uy += 0x7FFFu + ((uy >> 16) & 1u);
    // bytes: out = { uy[3], uy[2], ux[3], ux[2] }  (S0=uy -> sel 4..7, S1=ux -> sel 0..3)
    return __builtin_amdgcn_perm(uy, ux, 0x07060302u);
}

#define BM 128
#define BN 256
#define BK 32
#define LDS_STRIDE 40   // elements; 80B rows -> 16B-aligned, staggers banks

// C[M,N] = scale * (A[M,K] x B) + bias, bf16 WMMA, f32 accumulate.
template <bool A_F32, bool B_F32, bool B_NT, bool C_BF16>
__global__ void __launch_bounds__(256)
gemm_wmma(const void* __restrict__ Aptr, long long aBatch,
          const void* __restrict__ Bptr, long long bBatch,
          void* __restrict__ Cptr, long long cBatch,
          const float* __restrict__ bias, float outScale,
          int M, int N, int K)
{
    __shared__ unsigned short ldsA[BM * LDS_STRIDE];
    __shared__ unsigned short ldsB[BN * LDS_STRIDE];

    const int tid  = threadIdx.x;
    const int lane = tid & 31;
    const int w    = tid >> 5;          // 8 waves
    const int wm   = w >> 2;            // 0..1  -> 64-row slab
    const int wn   = w & 3;             // 0..3  -> 64-col slab
    const int rowW = wm * 64;
    const int colW = wn * 64;
    const int laneRow = lane & 15;
    const int hiHalf  = lane >> 4;      // 0/1

    const int m0 = blockIdx.y * BM;
    const int n0 = blockIdx.x * BN;
    const int bz = blockIdx.z;

    const float*          Af = ((const float*)Aptr)          + (size_t)bz * aBatch;
    const unsigned short* Ab = ((const unsigned short*)Aptr) + (size_t)bz * aBatch;
    const float*          Bf = ((const float*)Bptr)          + (size_t)bz * bBatch;
    const unsigned short* Bb = ((const unsigned short*)Bptr) + (size_t)bz * bBatch;

    FragC c[4][4];
#pragma unroll
    for (int i = 0; i < 4; ++i)
#pragma unroll
        for (int j = 0; j < 4; ++j)
#pragma unroll
            for (int r = 0; r < 8; ++r) c[i][j].f[r] = 0.0f;

    // staging assignments (256 threads)
    const int arow = tid >> 1;            // 0..127, 16 contiguous k each
    const int akk  = (tid & 1) * 16;
    const int bn   = tid;                 // 0..255, all 32 k each

    for (int k0 = 0; k0 < K; k0 += BK) {
        // ---- stage A tile: ldsA[row][k] (bf16), 16B vector path ----
        {
            unsigned short* dst = &ldsA[arow * LDS_STRIDE + akk];
            size_t g = (size_t)(m0 + arow) * K + (k0 + akk);
            if (A_F32) {
#pragma unroll
                for (int q = 0; q < 2; ++q) {
                    float4 f0 = *(const float4*)(Af + g + q * 8);
                    float4 f1 = *(const float4*)(Af + g + q * 8 + 4);
                    uint4 o;
                    o.x = pack2bf(f0.x, f0.y); o.y = pack2bf(f0.z, f0.w);
                    o.z = pack2bf(f1.x, f1.y); o.w = pack2bf(f1.z, f1.w);
                    *(uint4*)(dst + q * 8) = o;
                }
            } else {
#pragma unroll
                for (int q = 0; q < 2; ++q)
                    *(uint4*)(dst + q * 8) = *(const uint4*)(Ab + g + q * 8);
            }
        }
        // ---- stage B tile: ldsB[n][k] (bf16) ----
        {
            unsigned short* dst = &ldsB[bn * LDS_STRIDE];
            if (B_NT) {  // B[N,K]: 32 contiguous k -> 4x16B
                size_t g = (size_t)(n0 + bn) * K + k0;
                if (B_F32) {
#pragma unroll
                    for (int q = 0; q < 4; ++q) {
                        float4 f0 = *(const float4*)(Bf + g + q * 8);
                        float4 f1 = *(const float4*)(Bf + g + q * 8 + 4);
                        uint4 o;
                        o.x = pack2bf(f0.x, f0.y); o.y = pack2bf(f0.z, f0.w);
                        o.z = pack2bf(f1.x, f1.y); o.w = pack2bf(f1.z, f1.w);
                        *(uint4*)(dst + q * 8) = o;
                    }
                } else {
#pragma unroll
                    for (int q = 0; q < 4; ++q)
                        *(uint4*)(dst + q * 8) = *(const uint4*)(Bb + g + q * 8);
                }
            } else {      // B[K,N]: stride N along k (coalesced across threads)
                size_t g = (size_t)k0 * N + (n0 + bn);
                if (B_F32) {
#pragma unroll
                    for (int kk = 0; kk < BK; ++kk)
                        dst[kk] = f2bf(Bf[g + (size_t)kk * N]);
                } else {
#pragma unroll
                    for (int kk = 0; kk < BK; ++kk)
                        dst[kk] = Bb[g + (size_t)kk * N];
                }
            }
        }
        __syncthreads();

        // prefetch next K slab while we compute (global_prefetch_b8)
        if (k0 + BK < K) {
            size_t ga = (size_t)(m0 + arow) * K + (k0 + BK + akk);
            if (A_F32) __builtin_prefetch(&Af[ga]);
            else       __builtin_prefetch(&Ab[ga]);
            if (B_NT) {
                size_t gb = (size_t)(n0 + bn) * K + (k0 + BK);
                if (B_F32) __builtin_prefetch(&Bf[gb]);
                else       __builtin_prefetch(&Bb[gb]);
            } else {
                size_t gb = (size_t)(k0 + BK) * N + (n0 + bn);
                if (B_F32) __builtin_prefetch(&Bf[gb]);
                else       __builtin_prefetch(&Bb[gb]);
            }
        }

        // ---- build fragments per documented 16-bit WMMA layouts ----
        Frag16 a[4], b[4];
#pragma unroll
        for (int i = 0; i < 4; ++i) {
            const unsigned short* base = &ldsA[(rowW + i * 16 + laneRow) * LDS_STRIDE];
#pragma unroll
            for (int p = 0; p < 8; ++p) {
                int kb = ((p < 4) ? 2 * p : 2 * p + 8) + hiHalf * 8;
                a[i].u[p] = *(const unsigned int*)&base[kb];
            }
        }
#pragma unroll
        for (int j = 0; j < 4; ++j) {
            const unsigned short* base = &ldsB[(colW + j * 16 + laneRow) * LDS_STRIDE];
#pragma unroll
            for (int p = 0; p < 8; ++p) {
                int kb = ((p < 4) ? 2 * p : 2 * p + 8) + hiHalf * 8;
                b[j].u[p] = *(const unsigned int*)&base[kb];
            }
        }
#pragma unroll
        for (int i = 0; i < 4; ++i)
#pragma unroll
            for (int j = 0; j < 4; ++j)
                c[i][j].v = __builtin_amdgcn_wmma_f32_16x16x32_bf16(
                    false, a[i].v, false, b[j].v, (short)0, c[i][j].v, false, false);
        __syncthreads();
    }

    // ---- epilogue: scale, bias, store ----
    float*          Cf = ((float*)Cptr)          + (size_t)bz * cBatch;
    unsigned short* Cb = ((unsigned short*)Cptr) + (size_t)bz * cBatch;
#pragma unroll
    for (int i = 0; i < 4; ++i) {
#pragma unroll
        for (int j = 0; j < 4; ++j) {
            int col = n0 + colW + j * 16 + laneRow;
            float bv = bias ? bias[col] : 0.0f;
#pragma unroll
            for (int r = 0; r < 8; ++r) {
                int row = m0 + rowW + i * 16 + r + hiHalf * 8;
                float val = c[i][j].f[r] * outScale + bv;
                if (C_BF16) Cb[(size_t)row * N + col] = f2bf(val);
                else        Cf[(size_t)row * N + col] = val;
            }
        }
    }
}

// One block (256 threads) per score row: softmax over 2048 + dropout(p=0.5).
// Deterministic hash RNG stands in for JAX threefry (not bit-exact).
__global__ void __launch_bounds__(256)
softmax_dropout(const float* __restrict__ S, unsigned short* __restrict__ P, int LK)
{
    __shared__ float red[256];
    const unsigned int row = blockIdx.x;
    const float* s = S + (size_t)row * LK;
    unsigned short* p = P + (size_t)row * LK;
    const int tid = threadIdx.x;

    float vals[8];
    float m = -3.4e38f;
#pragma unroll
    for (int j = 0; j < 8; ++j) {
        vals[j] = s[tid + 256 * j];          // coalesced
        m = fmaxf(m, vals[j]);
    }
    red[tid] = m; __syncthreads();
    for (int off = 128; off > 0; off >>= 1) {
        if (tid < off) red[tid] = fmaxf(red[tid], red[tid + off]);
        __syncthreads();
    }
    m = red[0]; __syncthreads();

    float sum = 0.0f;
#pragma unroll
    for (int j = 0; j < 8; ++j) { vals[j] = __expf(vals[j] - m); sum += vals[j]; }
    red[tid] = sum; __syncthreads();
    for (int off = 128; off > 0; off >>= 1) {
        if (tid < off) red[tid] += red[tid + off];
        __syncthreads();
    }
    float inv = 1.0f / red[0];

#pragma unroll
    for (int j = 0; j < 8; ++j) {
        unsigned int idx = row * (unsigned int)LK + (unsigned int)(tid + 256 * j);
        unsigned int h = idx * 2654435761u;
        h ^= h >> 16; h *= 2246822519u; h ^= h >> 13; h *= 3266489917u; h ^= h >> 16;
        float keep = (h & 1u) ? 2.0f : 0.0f;   // 1/(1-0.5) = 2
        p[tid + 256 * j] = f2bf(vals[j] * inv * keep);
    }
}

extern "C" void kernel_launch(void* const* d_in, const int* in_sizes, int n_in,
                              void* d_out, int out_size, void* d_ws, size_t ws_size,
                              hipStream_t stream) {
    (void)in_sizes; (void)n_in; (void)out_size; (void)ws_size;
    const float* query = (const float*)d_in[0];
    const float* key   = (const float*)d_in[1];
    const float* value = (const float*)d_in[2];
    const float* Wq    = (const float*)d_in[3];
    const float* bq    = (const float*)d_in[4];
    const float* Wk    = (const float*)d_in[5];
    const float* bk    = (const float*)d_in[6];
    const float* Wv    = (const float*)d_in[7];
    const float* bv    = (const float*)d_in[8];

    const int B = 4, L = 2048, D = 1024;
    const int M = B * L;                       // 8192 flattened rows
    const float SCALE = 8.0f;                  // sqrt(1024/16), multiplied per source

    char* ws = (char*)d_ws;
    size_t qkvBytes = (size_t)M * D * 2;       // bf16
    unsigned short* Qb = (unsigned short*)(ws);
    unsigned short* Kb = (unsigned short*)(ws + qkvBytes);
    unsigned short* Vb = (unsigned short*)(ws + 2 * qkvBytes);
    float*          S  = (float*)(ws + 3 * qkvBytes);
    unsigned short* P  = (unsigned short*)(ws + 3 * qkvBytes + (size_t)B * L * L * 4);

    dim3 blk(256);

    // 1) projections: [8192,1024] x [1024,1024] (NN, f32) + bias -> bf16
    dim3 gproj(D / BN, M / BM, 1);
    gemm_wmma<true, true, false, true><<<gproj, blk, 0, stream>>>(
        query, 0, Wq, 0, Qb, 0, bq, 1.0f, M, D, D);
    gemm_wmma<true, true, false, true><<<gproj, blk, 0, stream>>>(
        key, 0, Wk, 0, Kb, 0, bk, 1.0f, M, D, D);
    gemm_wmma<true, true, false, true><<<gproj, blk, 0, stream>>>(
        value, 0, Wv, 0, Vb, 0, bv, 1.0f, M, D, D);

    // 2) scores: S = 8 * Q K^T, per batch (NT, bf16) -> f32
    dim3 gs(L / BN, L / BM, B);
    gemm_wmma<false, false, true, false><<<gs, blk, 0, stream>>>(
        Qb, (long long)L * D, Kb, (long long)L * D,
        S, (long long)L * L, nullptr, SCALE, L, L, D);

    // 3) softmax + dropout -> bf16 P
    softmax_dropout<<<dim3(B * L), blk, 0, stream>>>(S, P, L);

    // 4) O = P V (NN, bf16) -> f32 out
    dim3 go(D / BN, L / BM, B);
    gemm_wmma<false, false, false, false><<<go, blk, 0, stream>>>(
        P, (long long)L * L, Vb, (long long)L * D,
        d_out, (long long)L * D, nullptr, 1.0f, L, D, L);
}
